// RPN_28793460752480
// MI455X (gfx1250) — matrix operand order repaired and checked by reading, not compiled, not run on hardware
//
#include <hip/hip_runtime.h>
#include <hip/hip_bf16.h>
#include <math.h>

// ---------------------------------------------------------------------------
// CDNA5 (gfx1250) RPN forward: 3x3 conv + heads as bf16 WMMA implicit GEMMs.
// Conv pipeline: LDS-staged shared A fragments + async-to-LDS double-buffered
// B fragments (ASYNCcnt-tracked), then top-k / delta decode / per-image NMS.
// ---------------------------------------------------------------------------

typedef __attribute__((ext_vector_type(16))) __bf16 v16bf;
typedef __attribute__((ext_vector_type(8)))  float  v8f;

#define NLVL 5
#define CC   256
#define TOT  65472          // per-image anchors across all levels
#define RTOT 3960           // per-image proposals after per-level top-k
#define RPAD 4096
#define POSTK 1000
#define IMG  512.0f
#define NMSTH 0.7f
#define SCL_CLAMP 4.135166556742356f   // log(1000/16)
#define NEG_INF (-__builtin_inff())

__device__ __forceinline__ __bf16 f2bf(float f) {
  unsigned u = __builtin_bit_cast(unsigned, f);
  unsigned r = u + 0x7FFFu + ((u >> 16) & 1u);     // RNE
  unsigned short h = (unsigned short)(r >> 16);
  return __builtin_bit_cast(__bf16, h);
}

// Async copy of one 32B half-pair (B fragment slice) global -> LDS.
// VDST = per-lane LDS byte address (flat-pointer low 32 bits, ISA 10.2),
// VADDR = 64-bit global address; inst offset applies to both sides.
__device__ __forceinline__ void async_copy_b32B(const __bf16* g, __bf16* l) {
  unsigned la = (unsigned)(size_t)l;
  unsigned long long ga = (unsigned long long)(size_t)g;
  asm volatile("global_load_async_to_lds_b128 %0, %1, off"
               :: "v"(la), "v"(ga) : "memory");
  asm volatile("global_load_async_to_lds_b128 %0, %1, off offset:16"
               :: "v"(la), "v"(ga) : "memory");
}

// ---------------------------------------------------------------------------
// Weight repack: conv_w [256][256][3][3] f32 -> per-fragment bf16 B layout.
// Fragment (rs, ct, ot): 32 lanes x 16 bf16.
//   lane 0..15 : col = lane,    K = e       (K local 0..15)
//   lane 16..31: col = lane-16, K = 16+e    (K local 16..31)
// ---------------------------------------------------------------------------
__global__ void repack_conv_w(const float* __restrict__ w, __bf16* __restrict__ wb) {
  int idx = blockIdx.x * blockDim.x + threadIdx.x;
  if (idx >= 9 * 8 * 16 * 32 * 16) return;
  int e    = idx & 15;
  int lane = (idx >> 4) & 31;
  int frag = idx >> 9;                 // (rs*8+ct)*16+ot
  int ot = frag & 15;
  int ct = (frag >> 4) & 7;
  int rs = frag >> 7;
  int oc = (ot << 4) + (lane & 15);
  int c  = ct * 32 + ((lane >> 4) << 4) + e;
  int r  = rs / 3, s = rs - 3 * r;
  wb[idx] = f2bf(w[(((size_t)oc * CC + c) * 3 + r) * 3 + s]);
}

// Head weights: cols 0..2 = cls anchors, 3..14 = reg (a*4+d), 15 = zero pad.
__global__ void repack_head_w(const float* __restrict__ cls_w,
                              const float* __restrict__ reg_w,
                              __bf16* __restrict__ wh) {
  int idx = blockIdx.x * blockDim.x + threadIdx.x;
  if (idx >= 8 * 32 * 16) return;
  int e    = idx & 15;
  int lane = (idx >> 4) & 31;
  int ct   = idx >> 9;
  int col  = lane & 15;
  int c    = ct * 32 + ((lane >> 4) << 4) + e;
  float v = 0.0f;
  if (col < 3)       v = cls_w[(size_t)col * CC + c];
  else if (col < 15) v = reg_w[(size_t)(col - 3) * CC + c];
  wh[idx] = f2bf(v);
}

// ---------------------------------------------------------------------------
// 3x3 conv + ReLU as implicit GEMM.
// Block = 512 threads (16 waves) = one 16-pixel tile x all 256 out-channels.
// Per step (9 taps x 8 K-blocks = 72): the block stages the shared 16x32 A
// fragment once in LDS; each wave's B fragment for step k+1 is async-copied
// into a private double-buffered LDS slot while WMMA k runs (ASYNCcnt).
// ---------------------------------------------------------------------------
__global__ __launch_bounds__(512)
void conv3x3_relu_wmma(const float* __restrict__ x, const __bf16* __restrict__ wb,
                       const float* __restrict__ bias, float* __restrict__ t,
                       int H, int W) {
  __shared__ __bf16 sA[32 * 16];               // shared A fragment, [lane][e]
  __shared__ __bf16 sB[16 * 2 * 32 * 16];      // per-wave double-buffered B
  const int tid  = threadIdx.x;
  const int lane = tid & 31;
  const int wave = tid >> 5;                   // 0..15 = out-channel tile
  const int pt   = blockIdx.x;                 // pixel tile
  const int HW   = H * W;
  const int m0   = pt << 4;

  // Fill-role decode: tid = fe*32 + flane (lane-fast => coalesced gathers).
  const int fe     = tid >> 5;                 // fragment element 0..15
  const int fmrow  = lane & 15;                // pixel row within tile
  const int fahalf = lane >> 4;                // K-half select
  const int fkl    = fahalf * 16 + fe;
  const int fm   = m0 + fmrow;
  const int fimg = fm / HW;
  const int fpix = fm - fimg * HW;
  const int fy   = fpix / W;
  const int fx   = fpix - fy * W;
  const size_t fbase = (size_t)fimg * CC * HW;

  // B fragment stream for this wave: frag(step) = (step*16 + wave) * 512 elems
  const __bf16* bsrc = wb + (size_t)wave * (32 * 16) + (size_t)lane * 16;
  __bf16* slot0 = sB + (size_t)(wave * 2 + 0) * (32 * 16) + (size_t)lane * 16;
  __bf16* slot1 = sB + (size_t)(wave * 2 + 1) * (32 * 16) + (size_t)lane * 16;

  // Prologue: start async copy of step 0's B fragment into buffer 0.
  async_copy_b32B(bsrc, slot0);

  v8f acc = {};
  for (int step = 0; step < 72; ++step) {
    const int rs = step >> 3;
    const int ct = step & 7;
    const int r = rs / 3, s = rs - 3 * r;
    const int yy = fy + r - 1, xx = fx + s - 1;
    const bool inb = (yy >= 0) && (yy < H) && (xx >= 0) && (xx < W);

    __syncthreads();                           // previous A fully consumed
    float v = 0.0f;
    if (inb) {
      const size_t c = (size_t)(ct * 32 + fkl);
      v = x[fbase + c * HW + (size_t)yy * W + xx];
    }
    sA[lane * 16 + fe] = f2bf(v);
    // Overlap: launch async copy of next step's B fragment (other buffer).
    if (step < 71)
      async_copy_b32B(bsrc + (size_t)(step + 1) * (16 * 32 * 16),
                      ((step + 1) & 1) ? slot1 : slot0);
    __syncthreads();                           // A fragment visible

    // Wait for this step's B copy: async loads retire in order, so <=2
    // outstanding (the ones just issued) implies the current pair landed.
    if (step < 71)
      asm volatile("s_wait_asynccnt 0x2" ::: "memory");
    else
      asm volatile("s_wait_asynccnt 0x0" ::: "memory");

    v16bf afrag = ((const v16bf*)sA)[lane];                    // 32B ds load
    const __bf16* cur = (step & 1) ? slot1 : slot0;
    v16bf bfrag = *(const v16bf*)cur;                          // 32B ds load
    acc = __builtin_amdgcn_wmma_f32_16x16x32_bf16(false, afrag, false, bfrag,
                                                  (short)0, acc, false, false);
  }
  // D layout: element rr, lane l -> M = m0+rr+8*(l>>4), oc col = l&15
  const int oc = (wave << 4) + (lane & 15);
  const float bv = bias[oc];
  const int hi = lane >> 4;
#pragma unroll
  for (int rr = 0; rr < 8; ++rr) {
    const int mm = m0 + rr + (hi << 3);
    const int im2 = mm / HW;
    const int pp = mm - im2 * HW;
    float vv = acc[rr] + bv;
    vv = vv > 0.0f ? vv : 0.0f;
    t[((size_t)im2 * CC + oc) * HW + pp] = vv;
  }
}

// ---------------------------------------------------------------------------
// Fused cls(3) + reg(12) 1x1 heads as one padded 16-col WMMA GEMM, K=256.
// Writes scores [(n, base + pix*3 + a)] and deltas [(..)*4+d] (h,w,a order).
// ---------------------------------------------------------------------------
__global__ __launch_bounds__(256)
void head_wmma(const float* __restrict__ t, const __bf16* __restrict__ wh,
               const float* __restrict__ clsb, const float* __restrict__ regb,
               float* __restrict__ scores, float* __restrict__ deltas,
               int H, int W, int lvl_base) {
  const int lane = threadIdx.x & 31;
  const int pt = blockIdx.x * (blockDim.x >> 5) + (threadIdx.x >> 5);
  const int HW = H * W;
  if (pt >= (2 * HW) >> 4) return;               // wave-uniform
  const int m0 = pt << 4;
  const int ahalf = lane >> 4;
  const int m = m0 + (lane & 15);
  const int img = m / HW;
  const int pix = m - img * HW;
  const float* abase = t + (size_t)img * CC * HW + pix;

  v8f acc = {};
  for (int ct = 0; ct < 8; ++ct) {
    v16bf bfrag = ((const v16bf*)wh)[ct * 32 + lane];
    v16bf afrag = {};
    const int c0 = ct * 32 + ahalf * 16;
#pragma unroll
    for (int e = 0; e < 8; ++e) afrag[e] = f2bf(abase[(size_t)(c0 + e) * HW]);
#pragma unroll
    for (int e = 0; e < 8; ++e)
      afrag[8 + e] = f2bf(abase[(size_t)(c0 + 8 + e) * HW]);
    acc = __builtin_amdgcn_wmma_f32_16x16x32_bf16(false, afrag, false, bfrag,
                                                  (short)0, acc, false, false);
  }
  const int col = lane & 15;
  const int hi = lane >> 4;
#pragma unroll
  for (int rr = 0; rr < 8; ++rr) {
    const int mm = m0 + rr + (hi << 3);
    const int im2 = mm / HW;
    const int pp = mm - im2 * HW;
    const float v = acc[rr];
    if (col < 3) {
      scores[(size_t)im2 * TOT + lvl_base + (size_t)pp * 3 + col] = v + clsb[col];
    } else if (col < 15) {
      const int j = col - 3;
      deltas[((size_t)im2 * TOT + lvl_base + (size_t)pp * 3 + (j >> 2)) * 4 +
             (j & 3)] = v + regb[j];
    }
  }
}

// ---------------------------------------------------------------------------
// Per (image, level) iterative top-k (argmax reduction) + delta decode + clip.
// Destroys the scores buffer (regenerated every launch).
// ---------------------------------------------------------------------------
__global__ __launch_bounds__(256)
void topk_apply(float* __restrict__ scores, const float* __restrict__ deltas,
                const float* __restrict__ anchors,
                float* __restrict__ pb, float* __restrict__ ps,
                int* __restrict__ plvl,
                int lvl, int len, int k, int lvl_base, int out_off) {
  const int n = blockIdx.x;
  const int tid = threadIdx.x;
  float* sc = scores + (size_t)n * TOT + lvl_base;
  const float* dl = deltas + ((size_t)n * TOT + lvl_base) * 4;
  __shared__ float sv[256];
  __shared__ int si[256];
  for (int tsel = 0; tsel < k; ++tsel) {
    float best = NEG_INF;
    int bi = 0x7fffffff;
    for (int i = tid; i < len; i += 256) {
      float v = sc[i];
      if (v > best || (v == best && i < bi)) { best = v; bi = i; }
    }
    sv[tid] = best; si[tid] = bi;
    __syncthreads();
    for (int off = 128; off > 0; off >>= 1) {
      if (tid < off) {
        float ov = sv[tid + off]; int oi = si[tid + off];
        if (ov > sv[tid] || (ov == sv[tid] && oi < si[tid])) {
          sv[tid] = ov; si[tid] = oi;
        }
      }
      __syncthreads();
    }
    if (tid == 0) {
      const int idx = si[0];
      const float s = sv[0];
      sc[idx] = NEG_INF;
      const float* an = anchors + (size_t)idx * 4;
      const float wa = an[2] - an[0], ha = an[3] - an[1];
      const float cxa = an[0] + 0.5f * wa, cya = an[1] + 0.5f * ha;
      const float* d = dl + (size_t)idx * 4;
      const float dw = fminf(d[2], SCL_CLAMP), dh = fminf(d[3], SCL_CLAMP);
      const float cx = d[0] * wa + cxa, cy = d[1] * ha + cya;
      const float pw = expf(dw) * wa, ph = expf(dh) * ha;
      float x1 = cx - 0.5f * pw, y1 = cy - 0.5f * ph;
      float x2 = cx + 0.5f * pw, y2 = cy + 0.5f * ph;
      x1 = fminf(fmaxf(x1, 0.0f), IMG);
      y1 = fminf(fmaxf(y1, 0.0f), IMG);
      x2 = fminf(fmaxf(x2, 0.0f), IMG);
      y2 = fminf(fmaxf(y2, 0.0f), IMG);
      const bool fin = __builtin_isfinite(x1) && __builtin_isfinite(y1) &&
                       __builtin_isfinite(x2) && __builtin_isfinite(y2) &&
                       __builtin_isfinite(s);
      const bool ne = (x2 - x1 > 0.0f) && (y2 - y1 > 0.0f);
      const size_t o = (size_t)n * RTOT + out_off + tsel;
      pb[o * 4 + 0] = x1; pb[o * 4 + 1] = y1;
      pb[o * 4 + 2] = x2; pb[o * 4 + 3] = y2;
      ps[o] = (fin && ne) ? s : NEG_INF;
      plvl[o] = lvl;
    }
    __threadfence_block();
    __syncthreads();
  }
}

// ---------------------------------------------------------------------------
// Per-image: bitonic sort (desc, stable by index) in LDS, greedy same-level
// NMS, then ordered emit of first POSTK kept proposals.
// ---------------------------------------------------------------------------
__global__ __launch_bounds__(256)
void sort_nms_emit(const float* __restrict__ pb, const float* __restrict__ ps,
                   const int* __restrict__ plvl, float* __restrict__ out) {
  const int n = blockIdx.x;
  const int tid = threadIdx.x;
  __shared__ float skey[RPAD];
  __shared__ int sval[RPAD];
  __shared__ int skeep[RPAD];
  for (int i = tid; i < RPAD; i += 256) {
    skey[i] = (i < RTOT) ? ps[(size_t)n * RTOT + i] : NEG_INF;
    sval[i] = i;
  }
  __syncthreads();
  for (int sz = 2; sz <= RPAD; sz <<= 1) {
    for (int st = sz >> 1; st > 0; st >>= 1) {
      for (int i = tid; i < (RPAD >> 1); i += 256) {
        const int lo = ((i / st) * (st << 1)) + (i % st);
        const int hi = lo + st;
        const bool desc = ((lo & sz) == 0);
        const float ka = skey[lo], kb = skey[hi];
        const int ia = sval[lo], ib = sval[hi];
        const bool aFirst = (ka > kb) || (ka == kb && ia < ib);
        if (aFirst != desc) {
          skey[lo] = kb; skey[hi] = ka;
          sval[lo] = ib; sval[hi] = ia;
        }
      }
      __syncthreads();
    }
  }
  for (int i = tid; i < RPAD; i += 256)
    skeep[i] = (i < RTOT) && (skey[i] > NEG_INF);
  __syncthreads();

  for (int i = 0; i < RTOT - 1; ++i) {
    if (skeep[i]) {                       // uniform across block
      const int oi = sval[i];
      const float* bi = pb + ((size_t)n * RTOT + oi) * 4;
      const float x1 = bi[0], y1 = bi[1], x2 = bi[2], y2 = bi[3];
      const int li = plvl[(size_t)n * RTOT + oi];
      const float ar = (x2 - x1) * (y2 - y1);
      for (int j = i + 1 + tid; j < RTOT; j += 256) {
        if (!skeep[j]) continue;
        const int oj = sval[j];
        if (plvl[(size_t)n * RTOT + oj] != li) continue;
        const float* bj = pb + ((size_t)n * RTOT + oj) * 4;
        const float bx1 = bj[0], by1 = bj[1], bx2 = bj[2], by2 = bj[3];
        const float iw = fmaxf(fminf(x2, bx2) - fmaxf(x1, bx1), 0.0f);
        const float ih = fmaxf(fminf(y2, by2) - fmaxf(y1, by1), 0.0f);
        const float inter = iw * ih;
        const float uni = ar + (bx2 - bx1) * (by2 - by1) - inter;
        if (inter / fmaxf(uni, 1e-9f) > NMSTH) skeep[j] = 0;
      }
    }
    __syncthreads();
  }
  if (tid == 0) {
    int cnt = 0;
    for (int i = 0; i < RTOT && cnt < POSTK; ++i) {
      if (skeep[i]) {
        const int oi = sval[i];
        const float* b = pb + ((size_t)n * RTOT + oi) * 4;
        float* ob = out + (size_t)n * POSTK * 4 + (size_t)cnt * 4;
        ob[0] = b[0]; ob[1] = b[1]; ob[2] = b[2]; ob[3] = b[3];
        out[(size_t)2 * POSTK * 4 + (size_t)n * POSTK + cnt] = skey[i];
        ++cnt;
      }
    }
  }
}

__global__ void zero_out(float* __restrict__ o, int nfl) {
  int i = blockIdx.x * blockDim.x + threadIdx.x;
  if (i < nfl) o[i] = 0.0f;
}

// ---------------------------------------------------------------------------
extern "C" void kernel_launch(void* const* d_in, const int* in_sizes, int n_in,
                              void* d_out, int out_size, void* d_ws,
                              size_t ws_size, hipStream_t stream) {
  const float* feat[NLVL];
  for (int i = 0; i < NLVL; ++i) feat[i] = (const float*)d_in[i];
  const float* conv_w = (const float*)d_in[5];
  const float* conv_b = (const float*)d_in[6];
  const float* cls_w  = (const float*)d_in[7];
  const float* cls_b  = (const float*)d_in[8];
  const float* reg_w  = (const float*)d_in[9];
  const float* reg_b  = (const float*)d_in[10];
  const float* anch[NLVL];
  for (int i = 0; i < NLVL; ++i) anch[i] = (const float*)d_in[11 + i];
  float* out = (float*)d_out;

  static const int HL[NLVL]   = {128, 64, 32, 16, 8};
  static const int KL[NLVL]   = {1000, 1000, 1000, 768, 192};
  static const int BASE[NLVL] = {0, 49152, 61440, 64512, 65280};
  static const int OFF[NLVL]  = {0, 1000, 2000, 3000, 3768};

  // workspace carve-out
  char* ws = (char*)d_ws;
  size_t off = 0;
  auto carve = [&](size_t bytes) -> void* {
    void* p = ws + off;
    off += (bytes + 255) & ~(size_t)255;
    return p;
  };
  __bf16* wb = (__bf16*)carve((size_t)2304 * 256 * 2);            // conv B frags
  __bf16* wh = (__bf16*)carve((size_t)8 * 32 * 16 * 2);           // head B frags
  float* t      = (float*)carve((size_t)2 * CC * 128 * 128 * 4);  // conv output (reused)
  float* scores = (float*)carve((size_t)2 * TOT * 4);
  float* deltas = (float*)carve((size_t)2 * TOT * 4 * 4);
  float* pb     = (float*)carve((size_t)2 * RTOT * 4 * 4);
  float* ps     = (float*)carve((size_t)2 * RTOT * 4);
  int*   plvl   = (int*)carve((size_t)2 * RTOT * 4);

  zero_out<<<(out_size + 255) / 256, 256, 0, stream>>>(out, out_size);
  if (off > ws_size) return;  // insufficient scratch: deterministic zero output

  repack_conv_w<<<(9 * 8 * 16 * 32 * 16 + 255) / 256, 256, 0, stream>>>(conv_w, wb);
  repack_head_w<<<(8 * 32 * 16 + 255) / 256, 256, 0, stream>>>(cls_w, reg_w, wh);

  for (int l = 0; l < NLVL; ++l) {
    const int H = HL[l];
    const int convBlocks = (2 * H * H) / 16;  // one block per 16-pixel tile
    const int headBlocks = (H * H) / 64;      // (2*H*H/16) tiles / 8 waves
    conv3x3_relu_wmma<<<convBlocks, 512, 0, stream>>>(feat[l], wb, conv_b, t, H, H);
    head_wmma<<<headBlocks, 256, 0, stream>>>(t, wh, cls_b, reg_b, scores, deltas,
                                              H, H, BASE[l]);
  }
  for (int l = 0; l < NLVL; ++l) {
    const int HWA = HL[l] * HL[l] * 3;
    topk_apply<<<2, 256, 0, stream>>>(scores, deltas, anch[l], pb, ps, plvl,
                                      l, HWA, KL[l], BASE[l], OFF[l]);
  }
  sort_nms_emit<<<2, 256, 0, stream>>>(pb, ps, plvl, out);
}
